// RotateNMS_18854906429935
// MI455X (gfx1250) — compile-verified, matrix operand order checked by reading, not run on hardware
//
#include <hip/hip_runtime.h>
#include <hip/hip_bf16.h>
#include <math.h>
#include <stdint.h>

typedef __attribute__((ext_vector_type(16))) _Float16 v16h;
typedef __attribute__((ext_vector_type(8)))  float    v8f;

#define NMS_THR      0.7f
#define FILT_MARGIN  0.02f          /* conservative slack for f16 rounding */
#define COORD_SCALE  (1.0f/1024.0f) /* normalize coords into f16-safe range */

// ---------------------------------------------------------------- rank -----
// Stable rank == argsort(-scores): r_i = #{j : s_j > s_i || (s_j==s_i && j<i)}
__global__ void rank_kernel(const float* __restrict__ scores,
                            int* __restrict__ order, int N) {
    __shared__ float sh[256];
    int i = blockIdx.x * 256 + threadIdx.x;
    float si = (i < N) ? scores[i] : -3.0e38f;
    int r = 0;
    for (int base = 0; base < N; base += 256) {
        int j = base + threadIdx.x;
        sh[threadIdx.x] = (j < N) ? scores[j] : -3.0e38f;
        __syncthreads();
        int lim = (N - base < 256) ? (N - base) : 256;
        for (int k = 0; k < lim; ++k) {
            float sj = sh[k];
            int jj = base + k;
            if (sj > si || (sj == si && jj < i)) ++r;
        }
        __syncthreads();
    }
    if (i < N) order[r] = i;
}

// ---------------------------------------------------------------- prep -----
// For sorted slot t: CCW corners, area, and f16 prefilter features.
// dot(featA_i, featB_j) == dist^2(i,j) - (r_i + r_j)^2   (normalized units)
__global__ void prep_kernel(const float* __restrict__ boxes,
                            const int* __restrict__ order,
                            float* __restrict__ corners,
                            float* __restrict__ area,
                            _Float16* __restrict__ featA,
                            _Float16* __restrict__ featB, int N) {
    int t = blockIdx.x * blockDim.x + threadIdx.x;
    if (t >= N) return;
    int b = order[t];
    float xc = boxes[b * 5 + 0], yc = boxes[b * 5 + 1];
    float w  = boxes[b * 5 + 2], h  = boxes[b * 5 + 3];
    float th = boxes[b * 5 + 4];
    float ang = th * 0.017453292519943295f;
    float cs = cosf(ang), sn = sinf(ang);
    float dx = 0.5f * w, dy = 0.5f * h;
    const float sx[4] = {-1.f, 1.f, 1.f, -1.f};
    const float sy[4] = {-1.f, -1.f, 1.f, 1.f};
#pragma unroll
    for (int k = 0; k < 4; ++k) {
        float lx = sx[k] * dx, ly = sy[k] * dy;
        corners[t * 8 + 2 * k]     = xc + lx * cs - ly * sn;
        corners[t * 8 + 2 * k + 1] = yc + lx * sn + ly * cs;
    }
    area[t] = w * h;
    float xn = xc * COORD_SCALE, yn = yc * COORD_SCALE;
    float rr = 0.5f * sqrtf(w * w + h * h) * COORD_SCALE;
    float q  = xn * xn + yn * yn - rr * rr;
    _Float16* fa = featA + t * 8;
    _Float16* fb = featB + t * 8;
    fa[0] = (_Float16)xn; fa[1] = (_Float16)yn; fa[2] = (_Float16)rr;
    fa[3] = (_Float16)q;  fa[4] = (_Float16)1.0f;
    fa[5] = fa[6] = fa[7] = (_Float16)0.0f;
    fb[0] = (_Float16)(-2.f * xn); fb[1] = (_Float16)(-2.f * yn);
    fb[2] = (_Float16)(-2.f * rr); fb[3] = (_Float16)1.0f;
    fb[4] = (_Float16)q;
    fb[5] = fb[6] = fb[7] = (_Float16)0.0f;
}

// ---------------------------------------------------------------- zero -----
__global__ void zero_kernel(unsigned* __restrict__ p, int n) {
    int i = blockIdx.x * blockDim.x + threadIdx.x;
    if (i < n) p[i] = 0u;
}

// ------------------------------------------------------- exact clipping ----
// Sutherland–Hodgman: clip CCW rect A by CCW rect B edges; returns area.
__device__ __forceinline__ float rect_inter_area(const float* A, const float* B) {
    float px[8], py[8];
#pragma unroll
    for (int k = 0; k < 4; ++k) { px[k] = A[2 * k]; py[k] = A[2 * k + 1]; }
    int cnt = 4;
#pragma unroll 1
    for (int e = 0; e < 4; ++e) {
        float ax = B[2 * e], ay = B[2 * e + 1];
        int e1 = (e + 1) & 3;
        float ex = B[2 * e1] - ax, ey = B[2 * e1 + 1] - ay;
        float qx[8], qy[8];
        int nc = 0;
#pragma unroll 1
        for (int k = 0; k < cnt; ++k) {
            int kn = (k + 1 == cnt) ? 0 : (k + 1);
            float cx = px[k], cy = py[k], nx = px[kn], ny = py[kn];
            float dc = ex * (cy - ay) - ey * (cx - ax);
            float dn = ex * (ny - ay) - ey * (nx - ax);
            bool ic = (dc >= 0.f), inn = (dn >= 0.f);
            if (ic) { qx[nc] = cx; qy[nc] = cy; ++nc; }
            if (ic != inn) {
                float den = dc - dn;
                float tt = (fabsf(den) > 1e-12f) ? (dc / den) : 0.f;
                qx[nc] = cx + tt * (nx - cx);
                qy[nc] = cy + tt * (ny - cy);
                ++nc;
            }
        }
        cnt = nc;
#pragma unroll
        for (int k = 0; k < 8; ++k) { if (k < cnt) { px[k] = qx[k]; py[k] = qy[k]; } }
        if (cnt == 0) break;
    }
    float a2 = 0.f;
#pragma unroll 1
    for (int k = 0; k < cnt; ++k) {
        int kn = (k + 1 == cnt) ? 0 : (k + 1);
        a2 += px[k] * py[kn] - px[kn] * py[k];
    }
    return fmaxf(0.5f * a2, 0.f);
}

// ------------------------------------------------- WMMA filter + IoU -------
// One wave32 per 16x16 tile of (sorted row, sorted col) pairs, upper triangle.
// Corner records for the 32 tile boxes are staged into LDS with the CDNA5
// async-to-LDS engine (ASYNCcnt), then the WMMA circle-filter selects the
// few pairs needing the exact Sutherland-Hodgman clip (read from LDS).
__global__ void __launch_bounds__(32)
iou_tiles_kernel(const float* __restrict__ corners,
                 const float* __restrict__ area,
                 const _Float16* __restrict__ featA,
                 const _Float16* __restrict__ featB,
                 unsigned* __restrict__ mask, int N, int nblk) {
    int ct = blockIdx.x;  // column tile
    int rt = blockIdx.y;  // row tile
    if (ct < rt) return;  // only j > i needed
    __shared__ __align__(16) float shc[32 * 8];  // [0..15]=row tile, [16..31]=col tile
    int rowBase = rt * 16, colBase = ct * 16;
    int lane = threadIdx.x;

    // ---- async stage: each lane copies one 32-byte corner record ----------
    {
        int box = (lane < 16) ? (rowBase + lane) : (colBase + (lane - 16));
        if (box >= N) box = N - 1;                 // keep address valid
        unsigned lds = (unsigned)(uintptr_t)&shc[lane * 8];
        unsigned long long g =
            (unsigned long long)(uintptr_t)(corners + (size_t)box * 8);
        asm volatile(
            "global_load_async_to_lds_b128 %0, %1, off\n\t"
            "global_load_async_to_lds_b128 %0, %1, off offset:16\n\t"
            "s_wait_asynccnt 0x0"
            :: "v"(lds), "v"(g) : "memory");
    }

    // ---- WMMA circumscribed-circle prefilter ------------------------------
    // A: 16x32 f16 — lanes 0-15 hold K0..7 in halves 0..7 (features in K0..4).
    // B: 32x16 f16 — lanes 0-15 = column N, K0..7 in halves 0..7 (feat K0..4).
    v16h a, b;
#pragma unroll
    for (int k = 0; k < 16; ++k) { a[k] = (_Float16)0.f; b[k] = (_Float16)0.f; }
    if (lane < 16) {
        int rr = rowBase + lane;
        if (rr < N) {
            const _Float16* p = featA + rr * 8;
#pragma unroll
            for (int k = 0; k < 8; ++k) a[k] = p[k];
        }
        int cc = colBase + lane;
        if (cc < N) {
            const _Float16* p = featB + cc * 8;
#pragma unroll
            for (int k = 0; k < 8; ++k) b[k] = p[k];
        }
    }
    v8f cacc;
#pragma unroll
    for (int k = 0; k < 8; ++k) cacc[k] = 0.f;
    // d[v] = dist^2 - (r_i + r_j)^2 for pair (row = rowBase+v+8*(lane/16),
    //                                         col = colBase + lane%16)
    v8f d = __builtin_amdgcn_wmma_f32_16x16x32_f16(
        false, a, false, b, (short)0, cacc, false, false);

    int n = lane & 15;
    int hs = lane >> 4;
    int col_g = colBase + n;
    if (rowBase < N) __builtin_prefetch(area + rowBase, 0, 0);

    const float* colC = &shc[(16 + n) * 8];
#pragma unroll 1
    for (int v = 0; v < 8; ++v) {
        int row_g = rowBase + v + 8 * hs;
        if (col_g > row_g && col_g < N && d[v] < FILT_MARGIN) {
            const float* rowC = &shc[(v + 8 * hs) * 8];
            float inter = rect_inter_area(rowC, colC);
            float uni = area[row_g] + area[col_g] - inter;
            float iou = inter / fmaxf(uni, 1e-9f);
            if (iou > NMS_THR)
                atomicOr(&mask[(size_t)row_g * nblk + (col_g >> 5)],
                         1u << (col_g & 31));
        }
    }
}

// ----------------------------------------------------- sequential greedy ---
// One wave32; removal bitset lives in 2 VGPRs/lane (64 u32 words total).
__global__ void __launch_bounds__(32)
nms_seq_kernel(const unsigned* __restrict__ mask, const int* __restrict__ order,
               int* __restrict__ out, int N, int nblk, int cap) {
    int lane = threadIdx.x;
    unsigned r0 = 0u, r1 = 0u;  // words [lane] and [lane+32]
    int kc = 0;
    for (int i = 0; i < N; ++i) {
        int blk = i >> 5;
        unsigned word = (blk < 32) ? __shfl(r0, blk, 32) : __shfl(r1, blk - 32, 32);
        if (((word >> (i & 31)) & 1u) == 0u) {
            if (lane == 0 && kc < cap) out[kc] = order[i];
            ++kc;
            const unsigned* mrow = mask + (size_t)i * nblk;
            if (lane < nblk)      r0 |= mrow[lane];
            if (lane + 32 < nblk) r1 |= mrow[lane + 32];
        }
    }
    for (int t = lane; t < cap; t += 32)
        if (t >= kc) out[t] = -1;
}

// ---------------------------------------------------------------- launch ---
extern "C" void kernel_launch(void* const* d_in, const int* in_sizes, int n_in,
                              void* d_out, int out_size, void* d_ws, size_t ws_size,
                              hipStream_t stream) {
    const float* boxes  = (const float*)d_in[0];
    const float* scores = (const float*)d_in[1];
    int N = in_sizes[0] / 5;
    int nblk = (N + 31) / 32;

    char* ws = (char*)d_ws;
    size_t off = 0;
    auto alloc = [&](size_t bytes) -> void* {
        void* p = ws + off;
        off = (off + bytes + 255) & ~(size_t)255;
        return p;
    };
    int*      order   = (int*)alloc((size_t)N * 4);
    float*    area    = (float*)alloc((size_t)N * 4);
    float*    corners = (float*)alloc((size_t)N * 32);
    _Float16* featA   = (_Float16*)alloc((size_t)N * 16);
    _Float16* featB   = (_Float16*)alloc((size_t)N * 16);
    unsigned* mask    = (unsigned*)alloc((size_t)N * nblk * 4);
    (void)ws_size; (void)n_in;

    int* out = (int*)d_out;  // reference output dtype is int32

    rank_kernel<<<(N + 255) / 256, 256, 0, stream>>>(scores, order, N);
    prep_kernel<<<(N + 255) / 256, 256, 0, stream>>>(boxes, order, corners, area,
                                                     featA, featB, N);
    int total = N * nblk;
    zero_kernel<<<(total + 255) / 256, 256, 0, stream>>>(mask, total);

    int nT = (N + 15) / 16;
    dim3 grid(nT, nT);
    iou_tiles_kernel<<<grid, 32, 0, stream>>>(corners, area, featA, featB,
                                              mask, N, nblk);
    nms_seq_kernel<<<1, 32, 0, stream>>>(mask, order, out, N, nblk, out_size);
}
